// MetaSR_14886356648430
// MI455X (gfx1250) — compile-verified, matrix operand order checked by reading, not run on hardware
//
#include <hip/hip_runtime.h>
#include <hip/hip_bf16.h>

typedef _Float16 half_t;
typedef __attribute__((ext_vector_type(8)))  _Float16 v8h;
typedef __attribute__((ext_vector_type(16))) _Float16 v16h;
typedef __attribute__((ext_vector_type(8)))  float    v8f;

#define HID 256
#define KDIM 576
#define ST 264   // LDS row stride in halfs (16B-multiple, padded vs 256 to dodge bank conflicts)
#define MT 4     // M-tiles (of 16 queries) per wave -> 64 queries/wave

// A/B fragment slice loader (CDNA5 ISA 7.12.2 layout): p = row_base + kb + (lane>>4)*8 halfs.
// VGPR0-3 <- K = kh*8..kh*8+7 (packed pairs), VGPR4-7 <- K = 16+kh*8..16+kh*8+7
static __device__ __forceinline__ v16h ld_frag(const half_t* p) {
    v8h lo = *(const v8h*)p;
    v8h hi = *(const v8h*)(p + 16);
    return __builtin_shufflevector(lo, hi, 0,1,2,3,4,5,6,7,8,9,10,11,12,13,14,15);
}

// ---- Prep: transpose + f32->f16 convert w2 [256,256] -> w2T [256,256], w3 [256,576] -> w3T [576,256]
__global__ void metasr_prep(const float* __restrict__ w2, const float* __restrict__ w3,
                            half_t* __restrict__ w2t, half_t* __restrict__ w3t) {
    int i = blockIdx.x * blockDim.x + threadIdx.x;
    if (i < 256 * 256) {                 // w2t[n*256 + j] = w2[j*256 + n]
        int n = i >> 8, j = i & 255;
        w2t[i] = (half_t)w2[j * 256 + n];
    }
    if (i < KDIM * 256) {                // w3t[k*256 + j] = w3[j*576 + k]
        int k = i >> 8, j = i & 255;
        w3t[i] = (half_t)w3[j * KDIM + k];
    }
}

// ---- Main: one wave (32 threads) per block; each wave owns 64 queries (4 x 16-row M-tiles).
// B fragments are loaded once per (nt, kb) and reused across the 4 M-tiles (4x less L2 weight traffic).
__global__ void __launch_bounds__(32)
metasr_main(const float* __restrict__ feat, const float* __restrict__ coord,
            const float* __restrict__ cell, const float* __restrict__ w1,
            const float* __restrict__ b1,  const float* __restrict__ b2v,
            const float* __restrict__ b3v, const half_t* __restrict__ w2t,
            const half_t* __restrict__ w3t, float* __restrict__ out) {
    __shared__ half_t sh1[MT * 16 * ST];   // h1: 64 x 256 f16 (strided)
    __shared__ half_t sh2[MT * 16 * ST];   // h2: 64 x 256 f16 (strided)
    __shared__ float  sx[3][MT * 16];
    __shared__ int    siy[MT * 16], six[MT * 16];
    __shared__ float  spred[MT * 16];

    const int lane = threadIdx.x & 31;
    const int nl = lane & 15;     // M/N index within a 16-tile
    const int kh = lane >> 4;     // K-half selector
    const int q0 = blockIdx.x * (MT * 16);
    const int b  = q0 >> 13;      // Q = 8192, 64 | 8192 so one batch per block
    const int qi = q0 & 8191;

    // ---- Phase 1: per-query coordinate math (2 queries per lane)
    for (int m = lane; m < MT * 16; m += 32) {
        int q = qi + m;
        size_t base = ((size_t)b * 8192 + q) * 2;
        float cy = coord[base + 0], cx = coord[base + 1];
        float ly = cell[base + 0],  lx = cell[base + 1];
        float cy_ = cy - 0.5f * ly, cx_ = cx - 0.5f * lx;
        float cqy = fminf(fmaxf(cy_ + 1e-6f, -1.f + 1e-6f), 1.f - 1e-6f);
        float cqx = fminf(fmaxf(cx_ + 1e-6f, -1.f + 1e-6f), 1.f - 1e-6f);
        int iy = (int)rintf(((cqy + 1.f) * 64.f - 1.f) * 0.5f);
        int ix = (int)rintf(((cqx + 1.f) * 64.f - 1.f) * 0.5f);
        iy = min(max(iy, 0), 63);  ix = min(max(ix, 0), 63);
        float qy = 2.f * (float)iy / 64.f - 1.f;
        float qx = 2.f * (float)ix / 64.f - 1.f;
        sx[0][m] = (cy_ - qy) * 32.f;   // rel_y
        sx[1][m] = (cx_ - qx) * 32.f;   // rel_x
        sx[2][m] = ly * 32.f;           // r_rev
        siy[m] = iy;  six[m] = ix;
        spred[m] = 0.f;
    }
    __syncthreads();   // single-wave workgroup: lowers to s_nop, just orders LDS

    // ---- Phase 2: layer 1 (K=3, scalar), relu, f16 into LDS h1[64][256]
    for (int m = 0; m < MT * 16; ++m) {
        float x0 = sx[0][m], x1 = sx[1][m], x2 = sx[2][m];
        for (int j = lane; j < HID; j += 32) {
            float v = fmaf(x0, w1[j], fmaf(x1, w1[HID + j], fmaf(x2, w1[2 * HID + j], b1[j])));
            sh1[m * ST + j] = (half_t)fmaxf(v, 0.f);
        }
    }
    __syncthreads();

    // ---- Phase 3: h2 = relu(h1 @ w2 + b2); each B fragment feeds 4 WMMAs (one per M-tile)
    for (int nt = 0; nt < 16; ++nt) {
        v8f acc[MT];
        #pragma unroll
        for (int t = 0; t < MT; ++t) acc[t] = (v8f){};
        const half_t* brow = w2t + (size_t)(nt * 16 + nl) * HID + kh * 8;
        for (int kb = 0; kb < HID; kb += 32) {
            v16h bm = ld_frag(brow + kb);
            #pragma unroll
            for (int t = 0; t < MT; ++t) {
                v16h a = ld_frag(&sh1[(t * 16 + nl) * ST + kh * 8 + kb]);
                acc[t] = __builtin_amdgcn_wmma_f32_16x16x32_f16(
                    false, a, false, bm, (short)0, acc[t], false, false);
            }
        }
        int j = nt * 16 + nl;
        float bb = b2v[j];
        #pragma unroll
        for (int t = 0; t < MT; ++t)
            #pragma unroll
            for (int r = 0; r < 8; ++r) {      // D layout: VGPR r -> M=r (+8 for lanes 16-31)
                int m = t * 16 + r + 8 * kh;
                sh2[m * ST + j] = (half_t)fmaxf(acc[t][r] + bb, 0.f);
            }
    }
    __syncthreads();

    // ---- Phase 4: wmat tiles (h2 @ w3 + b3) fused with gather + dot-product reduction
    const float* featB = feat + (size_t)b * (64 * 64 * 64);
    float part[MT][8];
    #pragma unroll
    for (int t = 0; t < MT; ++t)
        #pragma unroll
        for (int r = 0; r < 8; ++r) part[t][r] = 0.f;

    for (int kt = 0; kt < 36; ++kt) {
        v8f acc[MT];
        #pragma unroll
        for (int t = 0; t < MT; ++t) acc[t] = (v8f){};
        const half_t* brow = w3t + (size_t)(kt * 16 + nl) * HID + kh * 8;
        for (int kb = 0; kb < HID; kb += 32) {
            v16h bm = ld_frag(brow + kb);
            #pragma unroll
            for (int t = 0; t < MT; ++t) {
                v16h a = ld_frag(&sh2[(t * 16 + nl) * ST + kh * 8 + kb]);
                acc[t] = __builtin_amdgcn_wmma_f32_16x16x32_f16(
                    false, a, false, bm, (short)0, acc[t], false, false);
            }
        }
        int k = kt * 16 + nl;                 // unfolded index: k = c*9 + (i*3 + j)
        float bb = b3v[k];
        int c   = k / 9;
        int pos = k - 9 * c;
        int di  = pos / 3 - 1;
        int dj  = pos - 3 * (pos / 3) - 1;
        const float* fc = featB + (size_t)c * 4096;
        #pragma unroll
        for (int t = 0; t < MT; ++t)
            #pragma unroll
            for (int r = 0; r < 8; ++r) {
                int m = t * 16 + r + 8 * kh;
                int y = siy[m] + di;
                int x = six[m] + dj;
                float qf = ((unsigned)y < 64u && (unsigned)x < 64u) ? fc[y * 64 + x] : 0.f;
                part[t][r] = fmaf(acc[t][r] + bb, qf, part[t][r]);
            }
    }
    // cross-lane (over N columns) reduction per query row via LDS float atomics (ds_add_f32)
    #pragma unroll
    for (int t = 0; t < MT; ++t)
        #pragma unroll
        for (int r = 0; r < 8; ++r)
            atomicAdd(&spred[t * 16 + r + 8 * kh], part[t][r]);
    __syncthreads();

    for (int m = lane; m < MT * 16; m += 32)
        out[(size_t)b * 8192 + qi + m] = spred[m];
}

extern "C" void kernel_launch(void* const* d_in, const int* in_sizes, int n_in,
                              void* d_out, int out_size, void* d_ws, size_t ws_size,
                              hipStream_t stream) {
    const float* feat  = (const float*)d_in[0];
    const float* coord = (const float*)d_in[1];
    const float* cell  = (const float*)d_in[2];
    const float* w1    = (const float*)d_in[3];
    const float* b1    = (const float*)d_in[4];
    const float* w2    = (const float*)d_in[5];
    const float* b2    = (const float*)d_in[6];
    const float* w3    = (const float*)d_in[7];
    const float* b3    = (const float*)d_in[8];
    float* out = (float*)d_out;

    half_t* w2t = (half_t*)d_ws;                                       // 128 KB
    half_t* w3t = (half_t*)((char*)d_ws + 256 * 256 * sizeof(half_t)); // 288 KB

    metasr_prep<<<(KDIM * 256 + 255) / 256, 256, 0, stream>>>(w2, w3, w2t, w3t);

    // 131072 queries / 64 per wave = 2048 blocks x 32 threads (1 wave each)
    metasr_main<<<2048, 32, 0, stream>>>(feat, coord, cell, w1, b1, b2, b3,
                                         w2t, w3t, out);
}
